// DSF_Bern_I_61357902790933
// MI455X (gfx1250) — matrix operand order repaired and checked by reading, not compiled, not run on hardware
//
#include <hip/hip_runtime.h>
#include <hip/hip_bf16.h>

#define NNODES 4096
#define NEDGES 131072
#define INDIM  1024
#define HIDDIM 512
#define OUTDIM 128
#define PEIN   128
#define PEHID  64
#define KHOPS  10
#define PE_ALPHA 0.1f
#define PE_BETA  0.5f

typedef __attribute__((ext_vector_type(16))) __bf16 v16bf;
typedef __attribute__((ext_vector_type(8)))  float  v8f;
typedef __attribute__((ext_vector_type(4)))  unsigned int u32x4;
typedef __attribute__((ext_vector_type(8)))  int i32x8;
typedef __attribute__((ext_vector_type(4)))  int i32x4;

#if defined(__has_builtin)
#if __has_builtin(__builtin_amdgcn_tensor_load_to_lds) && __has_builtin(__builtin_amdgcn_s_wait_tensorcnt)
#define HAVE_TDM 1
#endif
#endif
#ifndef HAVE_TDM
#define HAVE_TDM 0
#endif

__device__ __forceinline__ float sigmoidf_(float x) { return 1.0f / (1.0f + __expf(-x)); }

#if HAVE_TDM
// TDM: DMA a rows x cols tile of f32 (row stride in elements) from global to LDS.
// D# per CDNA5 ISA ch.8: group0 = {count|flags, lds_addr, global_addr, type=2},
// group1 = {data_size=4B, tensor_dim0/1, tile_dim0/1, tensor_dim0_stride}.
__device__ __forceinline__ void tdm_load_tile_f32(const float* gsrc, float* lds_dst,
                                                  unsigned rows, unsigned cols,
                                                  unsigned row_stride) {
    unsigned long long ga = (unsigned long long)(uintptr_t)gsrc;
    unsigned int laddr = (unsigned int)(uintptr_t)lds_dst;  // low 32 bits = LDS offset
    u32x4 g0;
    g0[0] = 1u;                                            // count=1, user descriptor
    g0[1] = laddr;                                         // lds_addr
    g0[2] = (unsigned int)(ga & 0xFFFFFFFFu);              // global_addr[31:0]
    g0[3] = (unsigned int)((ga >> 32) & 0x01FFFFFFu)       // global_addr[56:32]
            | (2u << 30);                                  // type = 2 (image)
    i32x8 g1;
    g1[0] = (int)(2u << 16);                               // data_size = 2 -> 4 bytes
    g1[1] = (int)((cols & 0xFFFFu) << 16);                 // tensor_dim0[15:0]
    g1[2] = (int)((cols >> 16) | ((rows & 0xFFFFu) << 16));// tensor_dim0[31:16] | tensor_dim1[15:0]
    g1[3] = (int)((cols & 0xFFFFu) << 16);                 // tile_dim0
    g1[4] = (int)(rows & 0xFFFFu);                         // tile_dim1 (tile_dim2 = 0)
    g1[5] = (int)row_stride;                               // tensor_dim0_stride[31:0]
    g1[6] = 0;                                             // stride hi | tensor_dim1_stride lo
    g1[7] = 0;
    i32x4 z4 = {0, 0, 0, 0};
#if __clang_major__ >= 23
    i32x8 z8 = {0, 0, 0, 0, 0, 0, 0, 0};
    __builtin_amdgcn_tensor_load_to_lds(g0, g1, z4, z4, z8, 0);
#else
    __builtin_amdgcn_tensor_load_to_lds(g0, g1, z4, z4, 0);
#endif
}
#endif

// ---------------------------------------------------------------- elementwise
__global__ void zero_kernel(float* __restrict__ p, long long n) {
    long long i = (long long)blockIdx.x * blockDim.x + threadIdx.x;
    if (i < n) p[i] = 0.0f;
}

__global__ void copy_kernel(float* __restrict__ dst, const float* __restrict__ src, long long n) {
    long long i = (long long)blockIdx.x * blockDim.x + threadIdx.x;
    if (i < n) dst[i] = src[i];
}

// ---------------------------------------------------------------- graph setup
__global__ void degree_kernel(const int* __restrict__ ei, float* __restrict__ deg,
                              float* __restrict__ deg2, int E) {
    int e = blockIdx.x * blockDim.x + threadIdx.x;
    if (e < E) {
        atomicAdd(&deg[ei[e]], 1.0f);       // src-degree
        atomicAdd(&deg2[ei[E + e]], 1.0f);  // dst-degree
    }
}

__global__ void edgew_kernel(const int* __restrict__ ei, const float* __restrict__ deg,
                             const float* __restrict__ deg2, float* __restrict__ wl,
                             float* __restrict__ wg, int E) {
    int e = blockIdx.x * blockDim.x + threadIdx.x;
    if (e >= E) return;
    int s = ei[e], d = ei[E + e];
    float ds = deg[s] > 0.0f ? rsqrtf(deg[s]) : 0.0f;
    float dd = deg[d] > 0.0f ? rsqrtf(deg[d]) : 0.0f;
    wl[e] = ds * dd;
    wg[e] = rsqrtf(deg2[s] + 1.0f) * rsqrtf(deg2[d] + 1.0f);
}

// out[dst] += sign * w[e] * x[src]  (out pre-initialized by caller)
__global__ void prop_kernel(float* __restrict__ out, const float* __restrict__ x,
                            const float* __restrict__ w, const int* __restrict__ ei,
                            int E, int F, float sign) {
    long long t = (long long)blockIdx.x * blockDim.x + threadIdx.x;
    int fchunks = F >> 2;
    long long total = (long long)E * fchunks;
    if (t >= total) return;
    int e  = (int)(t / fchunks);
    int f4 = (int)(t % fchunks) << 2;
    __builtin_prefetch(ei + e + 1024, 0, 1);  // global_prefetch_b8
    int s = ei[e], d = ei[E + e];
    float wv = sign * w[e];
    float4 xs = *reinterpret_cast<const float4*>(x + (long long)s * F + f4);
    float* o = out + (long long)d * F + f4;
    atomicAdd(o + 0, wv * xs.x);
    atomicAdd(o + 1, wv * xs.y);
    atomicAdd(o + 2, wv * xs.z);
    atomicAdd(o + 3, wv * xs.w);
}

// ---------------------------------------------------------------- WMMA GEMM
// C[M,N] = act(A[M,K] @ W[K,N] + bias[N]); act 0=none 1=relu 2=tanh
// One wave -> one 16x16 C tile, bf16 fragments, f32 accumulate.
__global__ void gemm_wmma_bias_act(const float* __restrict__ A, const float* __restrict__ W,
                                   const float* __restrict__ bias, float* __restrict__ C,
                                   int M, int Nn, int Kd, int act) {
    int lane = threadIdx.x & 31;
    int wave = threadIdx.x >> 5;
    int ntile = Nn >> 4;
    int tile = blockIdx.x * (blockDim.x >> 5) + wave;
    if (tile >= (M >> 4) * ntile) return;   // wave-uniform guard (EXEC all-ones)
    int tm = (tile / ntile) << 4;
    int tn = (tile % ntile) << 4;
    int m     = lane & 15;
    int koffA = (lane >> 4) << 3;   // A: lanes 0-15 hold K {0..7,16..23}, 16-31 hold {8..15,24..31}
    int nc    = lane & 15;
    int koffB = (lane >> 4) << 3;   // B: lanes 0-15 hold K {0..7,16..23}, 16-31 hold {8..15,24..31}

    const float* arow = A + (long long)(tm + m) * Kd;
    v8f acc = {};
    for (int k0 = 0; k0 < Kd; k0 += 32) {
        // A fragment: two contiguous 8-float runs -> 4x global_load_b128
        float4 a0 = *reinterpret_cast<const float4*>(arow + k0 + koffA);
        float4 a1 = *reinterpret_cast<const float4*>(arow + k0 + koffA + 4);
        float4 a2 = *reinterpret_cast<const float4*>(arow + k0 + 16 + koffA);
        float4 a3 = *reinterpret_cast<const float4*>(arow + k0 + 16 + koffA + 4);
        v16bf a, b;
        a[0] = (__bf16)a0.x; a[1] = (__bf16)a0.y; a[2] = (__bf16)a0.z; a[3] = (__bf16)a0.w;
        a[4] = (__bf16)a1.x; a[5] = (__bf16)a1.y; a[6] = (__bf16)a1.z; a[7] = (__bf16)a1.w;
        a[8] = (__bf16)a2.x; a[9] = (__bf16)a2.y; a[10] = (__bf16)a2.z; a[11] = (__bf16)a2.w;
        a[12] = (__bf16)a3.x; a[13] = (__bf16)a3.y; a[14] = (__bf16)a3.z; a[15] = (__bf16)a3.w;
#pragma unroll
        for (int i = 0; i < 16; ++i)  // lane-coalesced column reads of W
            b[i] = (__bf16)W[(long long)(k0 + koffB + (i & 7) + ((i >> 3) << 4)) * Nn + (tn + nc)];
        acc = __builtin_amdgcn_wmma_f32_16x16x32_bf16(false, a, false, b, (short)0, acc,
                                                      false, false);
    }
#pragma unroll
    for (int r = 0; r < 8; ++r) {
        int row = tm + r + ((lane >> 4) << 3);
        int col = tn + (lane & 15);
        float v = acc[r] + bias[col];
        if (act == 1) v = fmaxf(v, 0.0f);
        else if (act == 2) v = tanhf(v);
        C[(long long)row * Nn + col] = v;
    }
}

__device__ __forceinline__ v16bf cvt16(float4 a0, float4 a1, float4 a2, float4 a3) {
    v16bf a;
    a[0] = (__bf16)a0.x; a[1] = (__bf16)a0.y; a[2] = (__bf16)a0.z; a[3] = (__bf16)a0.w;
    a[4] = (__bf16)a1.x; a[5] = (__bf16)a1.y; a[6] = (__bf16)a1.z; a[7] = (__bf16)a1.w;
    a[8] = (__bf16)a2.x; a[9] = (__bf16)a2.y; a[10] = (__bf16)a2.z; a[11] = (__bf16)a2.w;
    a[12] = (__bf16)a3.x; a[13] = (__bf16)a3.y; a[14] = (__bf16)a3.z; a[15] = (__bf16)a3.w;
    return a;
}

// ---------------------------------------------------------------- fused pe_corr
// pecorr = sigmoid(c @ c^T) @ pe ; c,pe: [NNODES, 64]. One wave per 16-row tile.
// Block = 8 waves; per j-strip the 32x64 c/pe tiles are staged once into LDS
// (TDM tensor_load_to_lds when available) and shared by all 8 waves.
// Grid must be exactly NNODES/16/8 blocks of 256 threads (no guards).
__global__ void corr_fused(const float* __restrict__ c, const float* __restrict__ pe,
                           float* __restrict__ pecorr) {
    __shared__ float sc[32 * PEHID];    // c[j0..j0+31, 0:64]
    __shared__ float spe[32 * PEHID];   // pe[j0..j0+31, 0:64]
    __shared__ float strans[8][16 * 32];
    int lane = threadIdx.x & 31;
    int wave = threadIdx.x >> 5;
    int tm = (blockIdx.x * 8 + wave) << 4;
    int m     = lane & 15;
    int koffA = (lane >> 4) << 4;
    int nc    = lane & 15;
    int koffB = (lane >> 4) << 4;

    // Preload A fragments of c rows [tm..tm+15] (K = 64 -> two bf16 fragments)
    const float* crow = c + (long long)(tm + m) * PEHID;
    v16bf ca0 = cvt16(*reinterpret_cast<const float4*>(crow + koffA),
                      *reinterpret_cast<const float4*>(crow + koffA + 4),
                      *reinterpret_cast<const float4*>(crow + koffA + 8),
                      *reinterpret_cast<const float4*>(crow + koffA + 12));
    v16bf ca1 = cvt16(*reinterpret_cast<const float4*>(crow + 32 + koffA),
                      *reinterpret_cast<const float4*>(crow + 32 + koffA + 4),
                      *reinterpret_cast<const float4*>(crow + 32 + koffA + 8),
                      *reinterpret_cast<const float4*>(crow + 32 + koffA + 12));

    v8f acc[4] = {v8f{}, v8f{}, v8f{}, v8f{}};

    for (int j0 = 0; j0 < NNODES; j0 += 32) {
#if HAVE_TDM
        if (wave == 0) {
            tdm_load_tile_f32(c + (long long)j0 * PEHID, sc, 32, PEHID, PEHID);
            tdm_load_tile_f32(pe + (long long)j0 * PEHID, spe, 32, PEHID, PEHID);
            __builtin_amdgcn_s_wait_tensorcnt(0);
        }
#else
        for (int idx = threadIdx.x; idx < (32 * PEHID) / 4; idx += 256) {
            reinterpret_cast<float4*>(sc)[idx] =
                reinterpret_cast<const float4*>(c + (long long)j0 * PEHID)[idx];
            reinterpret_cast<float4*>(spe)[idx] =
                reinterpret_cast<const float4*>(pe + (long long)j0 * PEHID)[idx];
        }
#endif
        __syncthreads();

        // S strip: S[m, j0+jj] = <c[tm+m,:], c[j0+jj,:]>, jj = 0..31
        v8f s0 = {}, s1 = {};
#pragma unroll
        for (int half = 0; half < 2; ++half) {
            const float* brow = sc + (long long)(16 * half + nc) * PEHID;  // B[k,n] = c[j,k]
            v16bf b0 = cvt16(*reinterpret_cast<const float4*>(brow + koffB),
                             *reinterpret_cast<const float4*>(brow + koffB + 4),
                             *reinterpret_cast<const float4*>(brow + koffB + 8),
                             *reinterpret_cast<const float4*>(brow + koffB + 12));
            v16bf b1 = cvt16(*reinterpret_cast<const float4*>(brow + 32 + koffB),
                             *reinterpret_cast<const float4*>(brow + 32 + koffB + 4),
                             *reinterpret_cast<const float4*>(brow + 32 + koffB + 8),
                             *reinterpret_cast<const float4*>(brow + 32 + koffB + 12));
            v8f* sp = half ? &s1 : &s0;
            *sp = __builtin_amdgcn_wmma_f32_16x16x32_bf16(false, ca0, false, b0, (short)0, *sp,
                                                          false, false);
            *sp = __builtin_amdgcn_wmma_f32_16x16x32_bf16(false, ca1, false, b1, (short)0, *sp,
                                                          false, false);
        }
        // sigmoid in registers, transpose strip through LDS (D layout -> A layout)
#pragma unroll
        for (int r = 0; r < 8; ++r) {
            int row = r + ((lane >> 4) << 3);
            int col = lane & 15;
            strans[wave][row * 32 + col]      = sigmoidf_(s0[r]);
            strans[wave][row * 32 + col + 16] = sigmoidf_(s1[r]);
        }
        __syncthreads();
        v16bf a2;
#pragma unroll
        for (int i = 0; i < 16; ++i) {
            int kk = koffA + i;
            a2[i] = (__bf16)strans[wave][m * 32 + kk];
        }
        // accumulate: pecorr[tm.., 0:64] += S_strip(16x32) @ pe[j0..j0+31, 0:64]
#pragma unroll
        for (int t = 0; t < 4; ++t) {
            v16bf pb;
#pragma unroll
            for (int i = 0; i < 16; ++i)
                pb[i] = (__bf16)spe[(koffB + i) * PEHID + t * 16 + nc];
            acc[t] = __builtin_amdgcn_wmma_f32_16x16x32_bf16(false, a2, false, pb, (short)0,
                                                             acc[t], false, false);
        }
        __syncthreads();  // protect sc/spe/strans before next strip
    }
#pragma unroll
    for (int t = 0; t < 4; ++t)
#pragma unroll
        for (int r = 0; r < 8; ++r)
            pecorr[(long long)(tm + r + ((lane >> 4) << 3)) * PEHID + t * 16 + (lane & 15)] =
                acc[t][r];
}

// ---------------------------------------------------------------- PE update
// pe = tanh(alpha*rawpe + (1-alpha)*((1+beta)*(petpo + d2^2*pe) - beta*pecorr))
__global__ void pe_update_kernel(float* __restrict__ pe, const float* __restrict__ rawpe,
                                 const float* __restrict__ petpo, const float* __restrict__ pecorr,
                                 const float* __restrict__ deg2, long long n) {
    long long i = (long long)blockIdx.x * blockDim.x + threadIdx.x;
    if (i >= n) return;
    int node = (int)(i >> 6);  // PEHID = 64
    float d2 = rsqrtf(deg2[node] + 1.0f);
    float tpo = petpo[i] + d2 * d2 * pe[i];
    float v = (1.0f + PE_BETA) * tpo - PE_BETA * pecorr[i];
    v = PE_ALPHA * rawpe[i] + (1.0f - PE_ALPHA) * v;
    pe[i] = tanhf(v);
}

// gamma[node] = relu(temp[row]) * sigmoid(<pe[node,:], gate_w[row,:]> + gate_b[row])
__global__ void gamma_kernel(const float* __restrict__ pe, const float* __restrict__ gate_w,
                             const float* __restrict__ gate_b, const float* __restrict__ temp,
                             int row, float* __restrict__ gamma) {
    int node = blockIdx.x * blockDim.x + threadIdx.x;
    if (node >= NNODES) return;
    float acc = gate_b[row];
#pragma unroll 8
    for (int f = 0; f < PEHID; ++f)
        acc += pe[(long long)node * PEHID + f] * gate_w[row * PEHID + f];
    gamma[node] = fmaxf(temp[row], 0.0f) * sigmoidf_(acc);
}

__global__ void out_update_kernel(float* __restrict__ out, const float* __restrict__ gamma,
                                  const float* __restrict__ xi, float coeff, long long n,
                                  int init) {
    long long i = (long long)blockIdx.x * blockDim.x + threadIdx.x;
    if (i >= n) return;
    int node = (int)(i >> 7);  // OUTDIM = 128
    float v = coeff * gamma[node] * xi[i];
    if (init) out[i] = v; else out[i] += v;
}

// ---------------------------------------------------------------- host
static inline int cdiv_ll(long long n, int b) { return (int)((n + b - 1) / b); }

extern "C" void kernel_launch(void* const* d_in, const int* in_sizes, int n_in,
                              void* d_out, int out_size, void* d_ws, size_t ws_size,
                              hipStream_t stream) {
    const float* node_feat = (const float*)d_in[0];
    const int*   ei        = (const int*)d_in[1];
    const float* pos_enc   = (const float*)d_in[2];
    const float* lin1_w    = (const float*)d_in[3];
    const float* lin1_b    = (const float*)d_in[4];
    const float* lin2_w    = (const float*)d_in[5];
    const float* lin2_b    = (const float*)d_in[6];
    const float* pe_w      = (const float*)d_in[7];
    const float* pe_b      = (const float*)d_in[8];
    const float* cor_w     = (const float*)d_in[9];
    const float* cor_b     = (const float*)d_in[10];
    const float* gate_w    = (const float*)d_in[11];
    const float* gate_b    = (const float*)d_in[12];
    const float* temp      = (const float*)d_in[13];

    float* ws = (float*)d_ws;
    size_t off = 0;
    auto alloc = [&](size_t n) { float* p = ws + off; off += n; return p; };
    float* deg    = alloc(NNODES);
    float* deg2   = alloc(NNODES);
    float* wl     = alloc(NEDGES);
    float* wg     = alloc(NEDGES);
    float* x1     = alloc((size_t)NNODES * HIDDIM);
    float* tmp    = alloc((size_t)(KHOPS + 1) * NNODES * OUTDIM);
    float* xiA    = alloc((size_t)NNODES * OUTDIM);
    float* xiB    = alloc((size_t)NNODES * OUTDIM);
    float* pe     = alloc((size_t)NNODES * PEHID);
    float* rawpe  = alloc((size_t)NNODES * PEHID);
    float* cbuf   = alloc((size_t)NNODES * PEHID);
    float* pecorr = alloc((size_t)NNODES * PEHID);
    float* petpo  = alloc((size_t)NNODES * PEHID);
    float* gamma  = alloc(NNODES);

    float* outp   = (float*)d_out;                                // [N, 128]
    float* pe_out = (float*)d_out + (size_t)NNODES * OUTDIM;      // [N, 64]

    const long long NO = (long long)NNODES * OUTDIM;
    const long long NP = (long long)NNODES * PEHID;

    // binomial coefficients C(K,i)/2^K
    float coeff[KHOPS + 1];
    {
        double c = 1.0;
        for (int i = 0; i <= KHOPS; ++i) {
            coeff[i] = (float)(c / 1024.0);
            c = c * (KHOPS - i) / (i + 1);
        }
    }

    auto gemm = [&](const float* A, const float* W, const float* B, float* C,
                    int M, int N, int K, int act) {
        int tiles = (M >> 4) * (N >> 4);
        gemm_wmma_bias_act<<<cdiv_ll(tiles, 8), 256, 0, stream>>>(A, W, B, C, M, N, K, act);
    };

    // degrees & edge weights
    zero_kernel<<<cdiv_ll(2 * NNODES, 256), 256, 0, stream>>>(deg, 2 * NNODES);
    degree_kernel<<<cdiv_ll(NEDGES, 256), 256, 0, stream>>>(ei, deg, deg2, NEDGES);
    edgew_kernel<<<cdiv_ll(NEDGES, 256), 256, 0, stream>>>(ei, deg, deg2, wl, wg, NEDGES);

    // encoders
    gemm(pos_enc, pe_w, pe_b, pe, NNODES, PEHID, PEIN, /*tanh*/2);
    copy_kernel<<<cdiv_ll(NP, 256), 256, 0, stream>>>(rawpe, pe, NP);
    gemm(node_feat, lin1_w, lin1_b, x1, NNODES, HIDDIM, INDIM, /*relu*/1);
    gemm(x1, lin2_w, lin2_b, tmp, NNODES, OUTDIM, HIDDIM, /*none*/0);

    // tmp[k] = (I + A_norm) tmp[k-1]
    long long propNO = (long long)NEDGES * (OUTDIM >> 2);
    long long propNP = (long long)NEDGES * (PEHID >> 2);
    for (int k = 1; k <= KHOPS; ++k) {
        float* dst = tmp + (size_t)k * NO;
        float* src = tmp + (size_t)(k - 1) * NO;
        copy_kernel<<<cdiv_ll(NO, 256), 256, 0, stream>>>(dst, src, NO);
        prop_kernel<<<cdiv_ll(propNO, 256), 256, 0, stream>>>(dst, src, wl, ei, NEDGES, OUTDIM, 1.0f);
    }

    // out = coeff[0] * gamma0 * tmp[K]
    gamma_kernel<<<cdiv_ll(NNODES, 256), 256, 0, stream>>>(pe, gate_w, gate_b, temp, 0, gamma);
    out_update_kernel<<<cdiv_ll(NO, 256), 256, 0, stream>>>(outp, gamma, tmp + (size_t)KHOPS * NO,
                                                            coeff[0], NO, 1);

    for (int i = 0; i < KHOPS; ++i) {
        // xi = L^(i+1) tmp[K-i-1]
        copy_kernel<<<cdiv_ll(NO, 256), 256, 0, stream>>>(xiA, tmp + (size_t)(KHOPS - i - 1) * NO, NO);
        for (int rep = 0; rep <= i; ++rep) {
            copy_kernel<<<cdiv_ll(NO, 256), 256, 0, stream>>>(xiB, xiA, NO);
            prop_kernel<<<cdiv_ll(propNO, 256), 256, 0, stream>>>(xiB, xiA, wl, ei, NEDGES, OUTDIM, -1.0f);
            float* t2 = xiA; xiA = xiB; xiB = t2;
        }
        // c = pe @ cor_w + cor_b ; pecorr = sigmoid(c c^T) pe  (fused, WMMA + TDM staging)
        gemm(pe, cor_w, cor_b, cbuf, NNODES, PEHID, PEHID, 0);
        corr_fused<<<NNODES / 16 / 8, 256, 0, stream>>>(cbuf, pe, pecorr);
        // petpo = scatter part of Ahat pe
        zero_kernel<<<cdiv_ll(NP, 256), 256, 0, stream>>>(petpo, NP);
        prop_kernel<<<cdiv_ll(propNP, 256), 256, 0, stream>>>(petpo, pe, wg, ei, NEDGES, PEHID, 1.0f);
        pe_update_kernel<<<cdiv_ll(NP, 256), 256, 0, stream>>>(pe, rawpe, petpo, pecorr, deg2, NP);
        // gated output accumulation
        gamma_kernel<<<cdiv_ll(NNODES, 256), 256, 0, stream>>>(pe, gate_w, gate_b, temp, i + 1, gamma);
        out_update_kernel<<<cdiv_ll(NO, 256), 256, 0, stream>>>(outp, gamma, xiA, coeff[i + 1], NO, 0);
    }

    copy_kernel<<<cdiv_ll(NP, 256), 256, 0, stream>>>(pe_out, pe, NP);
}
